// Enc_coor_11166914969919
// MI455X (gfx1250) — matrix operand order repaired and checked by reading, not compiled, not run on hardware
//
#include <hip/hip_runtime.h>
#include <math.h>

// Problem constants (from reference)
#define FPX   96
#define DPX   28
#define KSLOT 3
#define SDIM  4
#define BDIM  64
#define TDIM  8
#define HDIM  512
#define ZDIM  2
#define CPX   69                 // FP - DP + 1
#define NPX   (CPX * CPX)        // 4761
#define SB    (SDIM * BDIM)      // 256
#define KPAD  4768               // NPX padded up to a multiple of 32 (WMMA K step)

typedef __attribute__((ext_vector_type(16))) _Float16 v16h;
typedef __attribute__((ext_vector_type(8)))  _Float16 v8h;
typedef __attribute__((ext_vector_type(8)))  float    v8f;

// ---------------------------------------------------------------------------
// CDNA5 async memory->LDS path (ASYNCcnt-tracked, bypasses VGPRs).
// Guarded so toolchains without the builtins fall back to staged copies.
// Builtin prototype (from hipcc diagnostic): b128 variant takes
//   (v4i addrspace(1)*, v4i addrspace(3)*, imm int offset, imm int cpol)
// ---------------------------------------------------------------------------
#if defined(__HIP_DEVICE_COMPILE__) && \
    __has_builtin(__builtin_amdgcn_global_load_async_to_lds_b128) && \
    __has_builtin(__builtin_amdgcn_s_wait_asynccnt)
#define USE_ASYNC_LDS 1
#else
#define USE_ASYNC_LDS 0
#endif

#if USE_ASYNC_LDS
typedef int v4i __attribute__((vector_size(4 * sizeof(int))));
typedef __attribute__((address_space(1))) v4i gv4i;
typedef __attribute__((address_space(3))) v4i lv4i;

__device__ __forceinline__ void async_copy_b128(const void* g, void* l)
{
    // global_load_async_to_lds_b128: per-lane 16B memory -> LDS, ASYNCcnt++
    __builtin_amdgcn_global_load_async_to_lds_b128(
        (gv4i*)(g), (lv4i*)(l), /*imm offset=*/0, /*cpol=*/0);
}
__device__ __forceinline__ void async_wait0()
{
    __builtin_amdgcn_s_wait_asynccnt(0);   // s_wait_asynccnt 0
}
#endif

// ---------------------------------------------------------------------------
// Kernel 0: frame_left = frames[:, :, timestep]
// ---------------------------------------------------------------------------
__global__ __launch_bounds__(256) void init_frame_kernel(
    const float* __restrict__ frames, const int* __restrict__ tstep,
    float* __restrict__ frame_left)
{
    const int t  = threadIdx.x;
    const int sb = blockIdx.x;
    const int ts = *tstep;
    const float* src = frames + ((size_t)sb * TDIM + (size_t)ts) * (FPX * FPX);
    float*       dst = frame_left + (size_t)sb * (FPX * FPX);
#pragma unroll
    for (int i = 0; i < (FPX * FPX) / 256; ++i)
        dst[t + i * 256] = src[t + i * 256];
}

// ---------------------------------------------------------------------------
// Kernel 1: W1 (4761x512 f32, k-major) -> W1t (512 x KPAD f16, n-major, zero pad)
// ---------------------------------------------------------------------------
__global__ __launch_bounds__(256) void w1_convert_kernel(
    const float* __restrict__ W1, _Float16* __restrict__ w1t)
{
    const int idx = blockIdx.x * 256 + threadIdx.x;   // over HDIM*KPAD
    const int n  = idx / KPAD;
    const int kp = idx - n * KPAD;
    w1t[(size_t)n * KPAD + kp] =
        (kp < NPX) ? (_Float16)W1[(size_t)kp * HDIM + n] : (_Float16)0.0f;
}

// ---------------------------------------------------------------------------
// Kernel 2: fused depthwise VALID correlation (96x96 * 28x28 -> 69x69)
//           + numerically-stable softmax, emitting f16 scores (K-padded).
//           One sample per workgroup; all intermediates in LDS.
// ---------------------------------------------------------------------------
__global__ __launch_bounds__(256) void conv_softmax_kernel(
    const float* __restrict__ frame_left, const float* __restrict__ ck,
    int k, _Float16* __restrict__ scores)
{
    __shared__ __align__(16) float sf[FPX * FPX];   // 36864 B
    __shared__ __align__(16) float sk[DPX * DPX];   //  3136 B
    __shared__ float sv[NPX];                       // 19044 B
    __shared__ float red[256];                      //  1024 B (~58.7 KB total)

    const int t  = threadIdx.x;
    const int sb = blockIdx.x;

    const float* fr = frame_left + (size_t)sb * (FPX * FPX);
    const float* kr = ck + ((size_t)sb * KSLOT + (size_t)k) * (DPX * DPX);

#if USE_ASYNC_LDS
    // 9216 floats = 2304 x 16B chunks = 9 rounds x 256 lanes
#pragma unroll
    for (int i = 0; i < (FPX * FPX) / (4 * 256); ++i) {
        const int c = i * 256 + t;                  // chunk index
        async_copy_b128(fr + 4 * c, &sf[4 * c]);
    }
    if (t < (DPX * DPX) / 4)                        // 196 chunks
        async_copy_b128(kr + 4 * t, &sk[4 * t]);
    async_wait0();                                  // drain ASYNCcnt before barrier
#else
#pragma unroll
    for (int i = 0; i < (FPX * FPX) / 256; ++i)
        sf[t + i * 256] = fr[t + i * 256];
    for (int j = t; j < DPX * DPX; j += 256)
        sk[j] = kr[j];
#endif
    __syncthreads();

    // correlation + local max
    float lmax = -1e30f;
    for (int idx = t; idx < NPX; idx += 256) {
        const int oy = idx / CPX;
        const int ox = idx - oy * CPX;
        const float* frow = &sf[oy * FPX + ox];
        float acc = 0.0f;
        for (int ky = 0; ky < DPX; ++ky) {
            const float* r  = frow + ky * FPX;
            const float* kw = &sk[ky * DPX];
#pragma unroll 7
            for (int kx = 0; kx < DPX; ++kx)
                acc = fmaf(r[kx], kw[kx], acc);
        }
        sv[idx] = acc;
        lmax = fmaxf(lmax, acc);
    }

    // block max
    red[t] = lmax; __syncthreads();
    for (int s2 = 128; s2 > 0; s2 >>= 1) {
        if (t < s2) red[t] = fmaxf(red[t], red[t + s2]);
        __syncthreads();
    }
    const float mx = red[0]; __syncthreads();

    // exp + block sum
    float lsum = 0.0f;
    for (int idx = t; idx < NPX; idx += 256) {
        const float e = expf(sv[idx] - mx);
        sv[idx] = e;
        lsum += e;
    }
    red[t] = lsum; __syncthreads();
    for (int s2 = 128; s2 > 0; s2 >>= 1) {
        if (t < s2) red[t] += red[t + s2];
        __syncthreads();
    }
    const float inv = 1.0f / red[0];

    _Float16* srow = scores + (size_t)sb * KPAD;
    for (int idx = t; idx < NPX; idx += 256)
        srow[idx] = (_Float16)(sv[idx] * inv);
    if (t < KPAD - NPX)                       // zero the K padding
        srow[NPX + t] = (_Float16)0.0f;
}

// ---------------------------------------------------------------------------
// Kernel 3: WMMA GEMM  h = relu(scores @ W1 + b1)
//   A: scores f16 [SB][KPAD] row-major;  B: W1t f16 [HDIM][KPAD] (n-major).
//   One 16x16 tile per wave (v_wmma_f32_16x16x32_f16), 4 waves/block.
// ---------------------------------------------------------------------------
__device__ inline v16h load_frag16(const _Float16* p0, const _Float16* p1)
{
    const v8h lo = *(const v8h*)p0;
    const v8h hi = *(const v8h*)p1;
    v16h r;
#pragma unroll
    for (int i = 0; i < 8; ++i) { r[i] = lo[i]; r[i + 8] = hi[i]; }
    return r;
}

__global__ __launch_bounds__(128) void gemm_h_kernel(
    const _Float16* __restrict__ A,   // [SB][KPAD]
    const _Float16* __restrict__ Bt,  // [HDIM][KPAD]
    const float* __restrict__ b1,
    float* __restrict__ Hout)         // [SB][HDIM]
{
    const int lane = threadIdx.x & 31;
    const int wave = threadIdx.x >> 5;
    const int tile = blockIdx.x * 4 + wave;       // 16 m-tiles * 32 n-tiles = 512
    const int nt   = tile & 31;
    const int mt   = tile >> 5;
    const int lid  = lane & 15;
    const int half = lane >> 4;

    const int m = (mt << 4) + lid;                // A row this lane feeds
    const int n = (nt << 4) + lid;                // B column this lane feeds
    const _Float16* arow = A  + (size_t)m * KPAD;
    const _Float16* brow = Bt + (size_t)n * KPAD;

    v8f acc = {};
    for (int kk = 0; kk < KPAD; kk += 32) {
        if (kk + 32 < KPAD) {                     // uniform; lowers to global_prefetch
            __builtin_prefetch(arow + kk + 32, 0, 0);
            __builtin_prefetch(brow + kk + 32, 0, 0);
        }
        // A 16x32 f16 layout: lane half h holds K = h*8+{0..7} and 16+h*8+{0..7}
        const v16h af = load_frag16(arow + kk + half * 8,
                                    arow + kk + 16 + half * 8);
        // B 32x16 f16 layout: lane half h holds K = h*8+{0..7} and 16+h*8+{0..7}
        const v16h bf = load_frag16(brow + kk + half * 8,
                                    brow + kk + 16 + half * 8);
        acc = __builtin_amdgcn_wmma_f32_16x16x32_f16(
                  false, af, false, bf, (short)0, acc, false, false);
    }

    const float bias = b1[n];
#pragma unroll
    for (int v = 0; v < 8; ++v) {                 // C/D: m = v + 8*half, n = lane&15
        const int mrow = (mt << 4) + v + (half << 3);
        const float x = acc[v] + bias;
        Hout[(size_t)mrow * HDIM + n] = x > 0.0f ? x : 0.0f;
    }
}

// ---------------------------------------------------------------------------
// Kernel 4: mean/std heads + reparameterization; writes the 3 outputs + zbuf.
// ---------------------------------------------------------------------------
__global__ __launch_bounds__(256) void heads_kernel(
    const float* __restrict__ Hb,
    const float* __restrict__ Wm1, const float* __restrict__ bm1,
    const float* __restrict__ Wm2, const float* __restrict__ bm2,
    const float* __restrict__ Ws1, const float* __restrict__ bs1,
    const float* __restrict__ Ws2, const float* __restrict__ bs2,
    const float* __restrict__ eps, int k,
    float* __restrict__ out, float* __restrict__ zbuf)
{
    __shared__ __align__(16) float sh[HDIM];
    __shared__ float hm[256], hs[256], red[256], vals[4];
    const int t  = threadIdx.x;
    const int sb = blockIdx.x;

#if USE_ASYNC_LDS
    if (t < HDIM / 4)                               // 128 x 16B chunks
        async_copy_b128(Hb + (size_t)sb * HDIM + 4 * t, &sh[4 * t]);
    async_wait0();
#else
    sh[t]       = Hb[(size_t)sb * HDIM + t];
    sh[t + 256] = Hb[(size_t)sb * HDIM + t + 256];
#endif
    __syncthreads();

    float am = bm1[t], as_ = bs1[t];
    for (int i = 0; i < HDIM; ++i) {
        const float hv = sh[i];
        am  = fmaf(hv, Wm1[i * 256 + t], am);
        as_ = fmaf(hv, Ws1[i * 256 + t], as_);
    }
    hm[t] = am  > 0.0f ? am  : 0.0f;
    hs[t] = as_ > 0.0f ? as_ : 0.0f;
    __syncthreads();

    for (int d = 0; d < 2; ++d) {
        red[t] = hm[t] * Wm2[t * 2 + d]; __syncthreads();
        for (int s2 = 128; s2 > 0; s2 >>= 1) { if (t < s2) red[t] += red[t + s2]; __syncthreads(); }
        if (t == 0) vals[d] = red[0];
        __syncthreads();
        red[t] = hs[t] * Ws2[t * 2 + d]; __syncthreads();
        for (int s2 = 128; s2 > 0; s2 >>= 1) { if (t < s2) red[t] += red[t + s2]; __syncthreads(); }
        if (t == 0) vals[2 + d] = red[0];
        __syncthreads();
    }

    if (t < 2) {
        const float mean = tanhf(vals[t] + bm2[t]);
        const float stdv = expf(vals[2 + t] + bs2[t]);
        const float z    = mean + stdv * eps[sb * (KSLOT * ZDIM) + k * ZDIM + t];
        const int base   = sb * (KSLOT * ZDIM) + k * ZDIM + t;
        out[base]                            = mean;   // q_mean
        out[SB * KSLOT * ZDIM + base]        = stdv;   // q_std
        out[2 * SB * KSLOT * ZDIM + base]    = z;      // z_where
        zbuf[sb * 2 + t] = z;
    }
}

// ---------------------------------------------------------------------------
// Kernel 5: bilinear digit placement + frame_left -= recon
//   (torch grid_sample semantics per reference: zero pad outside [0,DP))
// ---------------------------------------------------------------------------
__global__ __launch_bounds__(256) void recon_kernel(
    const float* __restrict__ ck, const float* __restrict__ zbuf,
    int k, float* __restrict__ frame_left)
{
    __shared__ __align__(16) float sd[DPX * DPX];
    const int t  = threadIdx.x;
    const int sb = blockIdx.x;
    const float* kr = ck + ((size_t)sb * KSLOT + (size_t)k) * (DPX * DPX);
#if USE_ASYNC_LDS
    if (t < (DPX * DPX) / 4)                        // 196 x 16B chunks
        async_copy_b128(kr + 4 * t, &sd[4 * t]);
    async_wait0();
#else
    for (int j = t; j < DPX * DPX; j += 256) sd[j] = kr[j];
#endif
    __syncthreads();

    const float zx = zbuf[sb * 2 + 0];
    const float zy = zbuf[sb * 2 + 1];
    const float scale = (float)FPX / (float)DPX;
    float* fr = frame_left + (size_t)sb * (FPX * FPX);

    for (int p = t; p < FPX * FPX; p += 256) {
        const int i = p / FPX;
        const int j = p - i * FPX;
        const float ocx = (2.0f * j + 1.0f) / FPX - 1.0f;
        const float ocy = (2.0f * i + 1.0f) / FPX - 1.0f;
        const float px = ((scale * (ocx - zx) + 1.0f) * DPX - 1.0f) * 0.5f;
        const float py = ((scale * (ocy - zy) + 1.0f) * DPX - 1.0f) * 0.5f;
        const float fx0 = floorf(px), fy0 = floorf(py);
        const int x0 = (int)fx0, y0 = (int)fy0;
        const int x1 = x0 + 1,   y1 = y0 + 1;
        const float fx = px - fx0, fy = py - fy0;
        const float wx0 = (x0 >= 0 && x0 < DPX) ? (1.0f - fx) : 0.0f;
        const float wx1 = (x1 >= 0 && x1 < DPX) ? fx : 0.0f;
        const float wy0 = (y0 >= 0 && y0 < DPX) ? (1.0f - fy) : 0.0f;
        const float wy1 = (y1 >= 0 && y1 < DPX) ? fy : 0.0f;
        const int cx0 = min(max(x0, 0), DPX - 1), cx1 = min(max(x1, 0), DPX - 1);
        const int cy0 = min(max(y0, 0), DPX - 1), cy1 = min(max(y1, 0), DPX - 1);
        const float t0 = wy0 * sd[cy0 * DPX + cx0] + wy1 * sd[cy1 * DPX + cx0];
        const float t1 = wy0 * sd[cy0 * DPX + cx1] + wy1 * sd[cy1 * DPX + cx1];
        fr[p] -= wx0 * t0 + wx1 * t1;
    }
}

// ---------------------------------------------------------------------------
extern "C" void kernel_launch(void* const* d_in, const int* in_sizes, int n_in,
                              void* d_out, int out_size, void* d_ws, size_t ws_size,
                              hipStream_t stream)
{
    const float* frames = (const float*)d_in[0];
    const float* convk  = (const float*)d_in[1];
    const float* eps    = (const float*)d_in[2];
    const float* W1     = (const float*)d_in[3];
    const float* b1     = (const float*)d_in[4];
    const float* Wm1    = (const float*)d_in[5];
    const float* bm1    = (const float*)d_in[6];
    const float* Wm2    = (const float*)d_in[7];
    const float* bm2    = (const float*)d_in[8];
    const float* Ws1    = (const float*)d_in[9];
    const float* bs1    = (const float*)d_in[10];
    const float* Ws2    = (const float*)d_in[11];
    const float* bs2    = (const float*)d_in[12];
    const int*   tstep  = (const int*)d_in[13];
    float* out = (float*)d_out;

    // Workspace carve-out (~17.3 MB total)
    char* ws = (char*)d_ws;
    float*    frame_left = (float*)ws;     ws += (size_t)SB * FPX * FPX * sizeof(float);
    _Float16* w1t        = (_Float16*)ws;  ws += (size_t)HDIM * KPAD * sizeof(_Float16);
    _Float16* sc         = (_Float16*)ws;  ws += (size_t)SB * KPAD * sizeof(_Float16);
    float*    hbuf       = (float*)ws;     ws += (size_t)SB * HDIM * sizeof(float);
    float*    zbuf       = (float*)ws;

    init_frame_kernel<<<SB, 256, 0, stream>>>(frames, tstep, frame_left);
    w1_convert_kernel<<<(HDIM * KPAD) / 256, 256, 0, stream>>>(W1, w1t);

    for (int k = 0; k < KSLOT; ++k) {
        conv_softmax_kernel<<<SB, 256, 0, stream>>>(frame_left, convk, k, sc);
        gemm_h_kernel<<<(SB / 16) * (HDIM / 16) / 4, 128, 0, stream>>>(sc, w1t, b1, hbuf);
        heads_kernel<<<SB, 256, 0, stream>>>(hbuf, Wm1, bm1, Wm2, bm2,
                                             Ws1, bs1, Ws2, bs2, eps, k, out, zbuf);
        recon_kernel<<<SB, 256, 0, stream>>>(convk, zbuf, k, frame_left);
    }
}
